// ConLoss_4724464026088
// MI455X (gfx1250) — matrix operand order repaired and checked by reading, not compiled, not run on hardware
//
#include <hip/hip_runtime.h>

typedef _Float16 v16h __attribute__((ext_vector_type(16)));
typedef _Float16 v8h  __attribute__((ext_vector_type(8)));
typedef float    v8f  __attribute__((ext_vector_type(8)));

#define BS 16
#define N  1024
#define D  256
#define TAU_INV 100.0f
// 100 * log2(e): exp((d-1)*100) == exp2(d*C - C)
#define C_LOG2E 144.26950408889634f

// ---------------------------------------------------------------------------
// Kernel 1: L2-normalize x and y (fp32 math), emit f16 copies for WMMA, and
// pos[i] = (xn . yn) / tau  (computed in fp32 before f16 rounding).
// One wave (32 lanes) per vector pair; 8 floats per lane.
// ---------------------------------------------------------------------------
__global__ __launch_bounds__(256) void norm_kernel(const float* __restrict__ x,
                                                   const float* __restrict__ y,
                                                   _Float16* __restrict__ xh,
                                                   _Float16* __restrict__ yh,
                                                   float* __restrict__ pos) {
    int wid  = threadIdx.x >> 5;
    int lane = threadIdx.x & 31;
    int vec  = blockIdx.x * 8 + wid;          // < BS*N
    const float* xp = x + (size_t)vec * D + lane * 8;
    const float* yp = y + (size_t)vec * D + lane * 8;

    float xv[8], yv[8];
    float sxx = 0.f, syy = 0.f, sxy = 0.f;
#pragma unroll
    for (int i = 0; i < 8; ++i) {
        xv[i] = xp[i];
        yv[i] = yp[i];
        sxx += xv[i] * xv[i];
        syy += yv[i] * yv[i];
        sxy += xv[i] * yv[i];
    }
#pragma unroll
    for (int m = 16; m >= 1; m >>= 1) {
        sxx += __shfl_xor(sxx, m, 32);
        syy += __shfl_xor(syy, m, 32);
        sxy += __shfl_xor(sxy, m, 32);
    }
    float inx = 1.0f / fmaxf(sqrtf(sxx), 1e-12f);
    float iny = 1.0f / fmaxf(sqrtf(syy), 1e-12f);

    _Float16* xo = xh + (size_t)vec * D + lane * 8;
    _Float16* yo = yh + (size_t)vec * D + lane * 8;
#pragma unroll
    for (int i = 0; i < 8; ++i) {
        xo[i] = (_Float16)(xv[i] * inx);
        yo[i] = (_Float16)(yv[i] * iny);
    }
    if (lane == 0) pos[vec] = sxy * inx * iny * TAU_INV;
}

// ---------------------------------------------------------------------------
// Kernel 2: fused gram + streaming logsumexp.
// Block = (batch b, anchor type t, 16-row block rb). 8 waves stream 128
// 16x16 column tiles (same-matrix with diag masked + cross-matrix with diag
// kept). Each tile = 8x v_wmma_f32_16x16x32_f16 over K=256 with the B tile
// double-buffered in registers (next tile's 16 b128 loads issued before the
// current WMMA chain, giving staggered s_wait_loadcnt). Epilogue is a single
// v_fma + v_exp per element: exp(s-100) = exp2(d*C - C), C = 100*log2(e);
// the fixed shift of 100 is valid since cosine similarities are bounded by 1.
// ---------------------------------------------------------------------------
__global__ __launch_bounds__(256) void lse_kernel(const _Float16* __restrict__ xh,
                                                  const _Float16* __restrict__ yh,
                                                  float* __restrict__ lse) {
    __shared__ float sums[8][16];

    int rb   = blockIdx.x & 63;        // row block within batch
    int t    = (blockIdx.x >> 6) & 1;  // 0: x-anchors, 1: y-anchors
    int b    = blockIdx.x >> 7;        // batch
    int wid  = threadIdx.x >> 5;
    int lane = threadIdx.x & 31;
    int r    = lane & 15;              // column within tile / row within A
    int h    = lane >> 4;              // half-wave (selects K sub-chunk / row set)

    const _Float16* anchor = (t == 0 ? xh : yh) + (size_t)b * N * D;
    const _Float16* T0     = (t == 0 ? xh : yh) + (size_t)b * N * D;  // skip diag
    const _Float16* T1     = (t == 0 ? yh : xh) + (size_t)b * N * D;  // keep diag
    int rowBase = rb * 16;

    // A fragments (16x32 f16 layout): lane r = row, half h picks K range
    // [k+16h, k+16h+16)  -> two contiguous 16B loads.
    v16h a[8];
#pragma unroll
    for (int f = 0; f < 8; ++f) {
        const _Float16* row = anchor + (size_t)(rowBase + r) * D + f * 32;
        v8h lo = *(const v8h*)(row + h * 16);
        v8h hi = *(const v8h*)(row + h * 16 + 8);
#pragma unroll
        for (int i = 0; i < 8; ++i) { a[f][i] = lo[i]; a[f][8 + i] = hi[i]; }
    }

    float acc[8];
#pragma unroll
    for (int v = 0; v < 8; ++v) acc[v] = 0.0f;

    // B (32x16) per-lane source pointer for a given tile index:
    // lane r = column n = target row; half h picks K range [k+16h, k+16h+16)
    // -> one contiguous 32B load per k-step.
    auto brow = [&](int tile) -> const _Float16* {
        const _Float16* T = (tile >> 6) ? T1 : T0;
        return T + (size_t)(((tile & 63) * 16) + r) * D + h * 16;
    };

    // Prime the double buffer with this wave's first tile.
    v16h bcur[8], bnxt[8];
    {
        const _Float16* p = brow(wid);
#pragma unroll
        for (int f = 0; f < 8; ++f) bcur[f] = *(const v16h*)(p + f * 32);
    }

    for (int tile = wid; tile < 128; tile += 8) {
        // Issue next tile's loads before consuming the current buffer so the
        // 8-WMMA chain overlaps the memory latency.
        int nt = tile + 8;
        if (nt < 128) {
            const _Float16* p = brow(nt);
#pragma unroll
            for (int f = 0; f < 8; ++f) bnxt[f] = *(const v16h*)(p + f * 32);
        }

        v8f c = {};
#pragma unroll
        for (int f = 0; f < 8; ++f) {
            c = __builtin_amdgcn_wmma_f32_16x16x32_f16(
                    false, a[f], false, bcur[f], (short)0, c, false, false);
        }

        // Diagonal can only occur in the same-matrix tile whose column block
        // equals the row block -- a wave-uniform condition. There,
        // gi == gj  <=>  r == 8h+v (lane-local constant).
        if (tile == rb) {   // mat==0 && (tile&63)==rb, since tile<64 => mat==0
#pragma unroll
            for (int v = 0; v < 8; ++v) {
                float e = __builtin_amdgcn_exp2f(
                    __builtin_fmaf(c[v], C_LOG2E, -C_LOG2E));
                if (r == 8 * h + v) e = 0.0f;
                acc[v] += e;
            }
        } else {
#pragma unroll
            for (int v = 0; v < 8; ++v)
                acc[v] += __builtin_amdgcn_exp2f(
                    __builtin_fmaf(c[v], C_LOG2E, -C_LOG2E));
        }

#pragma unroll
        for (int f = 0; f < 8; ++f) bcur[f] = bnxt[f];
    }

    // Reduce over the 16 columns held by each half-wave (stays within half).
#pragma unroll
    for (int v = 0; v < 8; ++v) {
#pragma unroll
        for (int m = 8; m >= 1; m >>= 1) acc[v] += __shfl_xor(acc[v], m, 32);
    }
    if (r == 0) {
#pragma unroll
        for (int v = 0; v < 8; ++v) sums[wid][8 * h + v] = acc[v];
    }
    __syncthreads();

    // Deterministic fixed-order cross-wave sum, then LSE = 100 + log(sum).
    if (threadIdx.x < 16) {
        float s = 0.0f;
#pragma unroll
        for (int w = 0; w < 8; ++w) s += sums[w][threadIdx.x];
        lse[(size_t)blockIdx.x * 16 + threadIdx.x] = TAU_INV + logf(s);
    }
}

// ---------------------------------------------------------------------------
// Kernel 3: deterministic final reduction.
// loss = (sum(LSE over 2*bs*n rows) - 2*sum(pos)) / (2*bs*n)
// ---------------------------------------------------------------------------
__global__ __launch_bounds__(256) void reduce_kernel(const float* __restrict__ lse,
                                                     const float* __restrict__ pos,
                                                     float* __restrict__ out) {
    __shared__ float sh[256];
    float local = 0.0f;
    for (int i = threadIdx.x; i < 2 * BS * N; i += 256) local += lse[i];
    for (int i = threadIdx.x; i < BS * N; i += 256) local -= 2.0f * pos[i];
    sh[threadIdx.x] = local;
    __syncthreads();
    for (int s = 128; s > 0; s >>= 1) {
        if (threadIdx.x < s) sh[threadIdx.x] += sh[threadIdx.x + s];
        __syncthreads();
    }
    if (threadIdx.x == 0) out[0] = sh[0] / (float)(2 * BS * N);
}

// ---------------------------------------------------------------------------
extern "C" void kernel_launch(void* const* d_in, const int* in_sizes, int n_in,
                              void* d_out, int out_size, void* d_ws, size_t ws_size,
                              hipStream_t stream) {
    const float* x = (const float*)d_in[0];
    const float* y = (const float*)d_in[1];

    char* ws = (char*)d_ws;
    const size_t elems = (size_t)BS * N * D;
    _Float16* xh  = (_Float16*)ws;                         // 8 MB
    _Float16* yh  = (_Float16*)(ws + elems * 2);           // 8 MB
    float*    pos = (float*)(ws + elems * 4);              // 64 KB
    float*    lse = (float*)(ws + elems * 4 + (size_t)BS * N * 4); // 128 KB

    norm_kernel<<<BS * N / 8, 256, 0, stream>>>(x, y, xh, yh, pos);
    lse_kernel<<<BS * 2 * (N / 16), 256, 0, stream>>>(xh, yh, lse);
    reduce_kernel<<<1, 256, 0, stream>>>(lse, pos, (float*)d_out);
}
